// GaussianUpsampling_27814208209600
// MI455X (gfx1250) — matrix operand order repaired and checked by reading, not compiled, not run on hardware
//
#include <hip/hip_runtime.h>
#include <hip/hip_bf16.h>
#include <stdint.h>

typedef __attribute__((ext_vector_type(16))) _Float16 v16h;
typedef __attribute__((ext_vector_type(8)))  float    v8f;

#define B_       16
#define T_TEXT   512
#define C_       256
#define T_FEATS  4096
#define DELTA_   0.1f
#define MBLOCK   32      // frames per block
#define NTHREADS 128     // 4 waves (wave32)

__global__ __launch_bounds__(NTHREADS)
void gauss_upsample_wmma(const float*   __restrict__ hs,
                         const float*   __restrict__ ds,
                         const uint8_t* __restrict__ h_masks,
                         const uint8_t* __restrict__ d_masks,
                         float*         __restrict__ out)
{
    __shared__ float    c_l[T_TEXT];          // gaussian centers
    __shared__ float    scan_part[NTHREADS];  // scan partials
    __shared__ float    red[MBLOCK][4];       // per-row partial max/sum
    __shared__ float    rowstat[MBLOCK];      // row max, then 1/rowsum
    __shared__ _Float16 pmat[MBLOCK][T_TEXT]; // normalized softmax weights (f16)

    const int b   = blockIdx.y;
    const int f0  = blockIdx.x * MBLOCK;
    const int tid = threadIdx.x;

    // ---------- 1) centers c[j] = cumsum(ds)[j] - 0.5*ds[j] ----------
    const float* dsb = ds + (size_t)b * T_TEXT;
    float v0 = dsb[tid*4+0], v1 = dsb[tid*4+1], v2 = dsb[tid*4+2], v3 = dsb[tid*4+3];
    scan_part[tid] = v0 + v1 + v2 + v3;
    __syncthreads();
    #pragma unroll
    for (int off = 1; off < NTHREADS; off <<= 1) {
        float add = (tid >= off) ? scan_part[tid - off] : 0.0f;
        __syncthreads();
        scan_part[tid] += add;
        __syncthreads();
    }
    float run = (tid == 0) ? 0.0f : scan_part[tid - 1];
    run += v0; c_l[tid*4+0] = run - 0.5f*v0;
    run += v1; c_l[tid*4+1] = run - 0.5f*v1;
    run += v2; c_l[tid*4+2] = run - 0.5f*v2;
    run += v3; c_l[tid*4+3] = run - 0.5f*v3;
    __syncthreads();

    // ---------- 2) softmax over T_text for 32 frame rows ----------
    const int   m    = tid >> 2;                 // row 0..31
    const int   cs   = (tid & 3) * (T_TEXT/4);   // 128-col chunk
    const int   f    = f0 + m;
    const float tval = h_masks[(size_t)b*T_FEATS + f] ? (float)f : 0.0f;
    const uint8_t* dmb = d_masks + (size_t)b * T_TEXT;

    float mx = -3.402823e38f;
    for (int j = cs; j < cs + T_TEXT/4; ++j) {
        if (dmb[j]) {
            float d = tval - c_l[j];
            mx = fmaxf(mx, -DELTA_*d*d);
        }
    }
    red[m][tid & 3] = mx;
    __syncthreads();
    if (tid < MBLOCK)
        rowstat[tid] = fmaxf(fmaxf(red[tid][0], red[tid][1]),
                             fmaxf(red[tid][2], red[tid][3]));
    __syncthreads();
    const float rowmax = rowstat[m];

    float sum = 0.0f;
    for (int j = cs; j < cs + T_TEXT/4; ++j) {
        float p = 0.0f;
        if (dmb[j]) {
            float d = tval - c_l[j];
            p = __expf(-DELTA_*d*d - rowmax);
        }
        sum += p;
        pmat[m][j] = (_Float16)p;
    }
    red[m][tid & 3] = sum;
    __syncthreads();
    if (tid < MBLOCK)
        rowstat[tid] = 1.0f / (red[tid][0] + red[tid][1] + red[tid][2] + red[tid][3]);
    __syncthreads();
    const float invs = rowstat[m];
    for (int j = cs; j < cs + T_TEXT/4; ++j)
        pmat[m][j] = (_Float16)((float)pmat[m][j] * invs);
    __syncthreads();

    // ---------- 3) WMMA: [32 x 512] @ [512 x 256] -> [32 x 256] ----------
    const int wave = tid >> 5;
    const int lane = tid & 31;
    const int lm   = lane & 15;   // A row within tile / B,C,D column within tile
    const int hi   = lane >> 4;   // half-wave select

    v8f zero = {};
    v8f acc[2][4];
    #pragma unroll
    for (int i = 0; i < 2; ++i)
        #pragma unroll
        for (int j = 0; j < 4; ++j) acc[i][j] = zero;

    const _Float16* pr0 = &pmat[lm][0];        // M-tile 0 row
    const _Float16* pr1 = &pmat[16 + lm][0];   // M-tile 1 row
    const float* hsb = hs + (size_t)b * T_TEXT * C_;
    const int nbase = wave * 64 + lm;          // this lane's channel within block

    for (int k0 = 0; k0 < T_TEXT; k0 += 32) {
        // A fragment: 16-bit A 16x32 layout. K = (h&7) + 16*(h>>3) + 8*hi
        v16h a0, a1;
        #pragma unroll
        for (int hh = 0; hh < 8; ++hh) {
            const int K = k0 + (hh & 3)*2 + ((hh >> 2) << 4) + (hi << 3);
            a0[2*hh]   = pr0[K];
            a0[2*hh+1] = pr0[K+1];
            a1[2*hh]   = pr1[K];
            a1[2*hh+1] = pr1[K+1];
        }
        #pragma unroll
        for (int nt = 0; nt < 4; ++nt) {
            // B fragment: 32x16 layout. K = (h&7) + 16*(h>>3) + 8*hi, N = lm (fixed per lane)
            const float* colp = hsb + (size_t)(k0 + hi*8) * C_ + nbase + nt*16;
            v16h bf;
            #pragma unroll
            for (int h = 0; h < 16; ++h)
                bf[h] = (_Float16)colp[((h & 7) + ((h >> 3) << 4)) * C_];
            acc[0][nt] = __builtin_amdgcn_wmma_f32_16x16x32_f16(
                false, a0, false, bf, (short)0, acc[0][nt], false, false);
            acc[1][nt] = __builtin_amdgcn_wmma_f32_16x16x32_f16(
                false, a1, false, bf, (short)0, acc[1][nt], false, false);
        }
    }

    // ---------- 4) store D: M = r + 8*hi, N = lm ----------
    float* obase = out + ((size_t)b * T_FEATS + f0 + hi*8) * C_ + wave*64 + lm;
    #pragma unroll
    for (int mt = 0; mt < 2; ++mt)
        #pragma unroll
        for (int nt = 0; nt < 4; ++nt)
            #pragma unroll
            for (int r = 0; r < 8; ++r)
                obase[(size_t)(mt*16 + r) * C_ + nt*16] = acc[mt][nt][r];
}

extern "C" void kernel_launch(void* const* d_in, const int* in_sizes, int n_in,
                              void* d_out, int out_size, void* d_ws, size_t ws_size,
                              hipStream_t stream) {
    (void)in_sizes; (void)n_in; (void)out_size; (void)d_ws; (void)ws_size;
    const float*   hs = (const float*)d_in[0];
    const float*   ds = (const float*)d_in[1];
    const uint8_t* hm = (const uint8_t*)d_in[2];
    const uint8_t* dm = (const uint8_t*)d_in[3];
    float* out = (float*)d_out;

    dim3 grid(T_FEATS / MBLOCK, B_);
    gauss_upsample_wmma<<<grid, NTHREADS, 0, stream>>>(hs, ds, hm, dm, out);
}
